// T5SLA2Attention_86131274154619
// MI455X (gfx1250) — matrix operand order repaired and checked by reading, not compiled, not run on hardware
//
#include <hip/hip_runtime.h>
#include <hip/hip_bf16.h>

// ---------------------------------------------------------------------------
// Problem constants (from reference)
// ---------------------------------------------------------------------------
constexpr int kB   = 2;     // batch
constexpr int kL   = 4096;  // sequence length
constexpr int kDM  = 512;   // model dim
constexpr int kH   = 8;     // heads
constexpr int kDH  = 64;    // head dim
constexpr int kBS  = 64;    // block size
constexpr int kNB  = 64;    // number of blocks (L / BS)
constexpr int kKPQ = 9;     // selected key blocks per query block

typedef __attribute__((ext_vector_type(16))) _Float16 v16h;
typedef __attribute__((ext_vector_type(2)))  _Float16 v2h;
typedef __attribute__((ext_vector_type(8)))  float    v8f;
typedef __attribute__((ext_vector_type(4)))  int      v4i;

// Async global->LDS copy path (guarded: falls back to plain loads if the
// toolchain doesn't declare the gfx1250 async builtins).
#if defined(__gfx1250__) && __has_builtin(__builtin_amdgcn_global_load_async_to_lds_b128) && __has_builtin(__builtin_amdgcn_s_wait_asynccnt)
#define ASYNC_LDS 1
typedef __attribute__((address_space(1))) v4i* gptr_v4i;
typedef __attribute__((address_space(3))) v4i* lptr_v4i;
#else
#define ASYNC_LDS 0
#endif

// ---------------------------------------------------------------------------
// WMMA helpers (CDNA5 wave32, 16x16x32 f16 -> f32)
// ---------------------------------------------------------------------------
__device__ inline v8f wmma_f16(v16h a, v16h b, v8f c) {
  return __builtin_amdgcn_wmma_f32_16x16x32_f16(
      /*neg_a=*/false, a, /*neg_b=*/false, b,
      /*c_mod=*/(short)0, c, /*reuse_a=*/false, /*reuse_b=*/false);
}

// Fragment-major LDS layout: each lane's 16-half operand is one contiguous
// 32-byte chunk -> v16h load lowers to 2x ds_load_b128.
//
// A (16x32 MxK, per ISA): lane = m + 16*((k>>3)&1); pair p = ((k&7)>>1)+4*(k>>4);
//                         slot = 2p + (k&1)
// B (32x16 KxN, per ISA): lane = n + 16*((k>>3)&1); slot = (k&7) + 8*(k>>4)
// Index helpers for a 64x64 operand split into [kstep 0..1][frag 0..3][lane][slot]:
__device__ inline int a_frag_index(int r, int c) {  // (row M, col K) of 64x64
  int ks = c >> 5, cc = c & 31;
  int f = r >> 4, m = r & 15;
  int lane = m + 16 * ((cc >> 3) & 1);
  int p = ((cc & 7) >> 1) + 4 * (cc >> 4);
  return (((ks * 4 + f) * 32) + lane) * 16 + 2 * p + (cc & 1);
}
__device__ inline int b_frag_index(int kdim, int n) {  // (row K, col N) of 64x64
  int ks = kdim >> 5, kk = kdim & 31;
  int g = n >> 4, nl = n & 15;
  int lane = nl + 16 * ((kk >> 3) & 1);
  return (((ks * 4 + g) * 32) + lane) * 16 + (kk & 7) + 8 * (kk >> 4);
}

// C/D store: 16x16 f32 tile; lanes 0-15 -> N=lane, M=vgpr; lanes 16-31 -> M=8+vgpr.
__device__ inline void store_c16x16(float* C, int ldc, v8f c) {
  int lane = threadIdx.x & 31;
  int n = lane & 15;
  int mb = (lane >> 4) * 8;
#pragma unroll
  for (int vg = 0; vg < 8; ++vg) C[(mb + vg) * ldc + n] = c[vg];
}

// Symmetric 8-bit fake-quantize (qmax=127, qmin=-128, matches reference)
__device__ inline float fq(float x, float scale) {
  float r = rintf(x / scale);
  r = fminf(fmaxf(r, -128.f), 127.f);
  return r * scale;
}

__device__ inline float phi_fn(float x) {  // elu(x)+1
  return x > 0.f ? x + 1.f : __expf(x);
}

// ---------------------------------------------------------------------------
// Kernel 1: fused Q/K/V projection  out = hidden @ W.T  (M=8192, N=3*512, K=512)
// 64x64 output tile per block; 8 waves in 4x2 layout; K stepped by 32.
// ---------------------------------------------------------------------------
__global__ __launch_bounds__(256) void qkv_proj_kernel(
    const float* __restrict__ hidden, const float* __restrict__ Wq,
    const float* __restrict__ Wk, const float* __restrict__ Wv,
    float* __restrict__ q, float* __restrict__ k, float* __restrict__ v) {
  __shared__ v16h Af[4][32];   // 4 M-strips of 16x32
  __shared__ v16h Bf[4][32];   // 4 N-groups of 32x16
  _Float16* af = (_Float16*)Af;
  _Float16* bf = (_Float16*)Bf;

  const int bx = blockIdx.x;
  const int by = blockIdx.y;
  const int tid = threadIdx.x;
  const int wave = tid >> 5, lane = tid & 31;
  const int wm = wave >> 1, wn = wave & 1;

  const int jbase = by * 64;
  const int sel   = jbase / kDM;       // 0:q 1:k 2:v
  const int jcol  = jbase % kDM;
  const float* W   = (sel == 0) ? Wq : (sel == 1 ? Wk : Wv);
  float*       out = (sel == 0) ? q  : (sel == 1 ? k  : v);

  v8f acc0 = {}; v8f acc1 = {};

  for (int k0 = 0; k0 < kDM; k0 += 32) {
    if (k0 + 32 < kDM)  // global_prefetch_b8 hint for next A tile
      __builtin_prefetch(&hidden[(size_t)(bx * 64 + (tid >> 2)) * kDM + k0 + 32], 0, 1);
    // A fill: 1024 even/odd K pairs -> ds_store_b32 each
#pragma unroll
    for (int e = 0; e < 4; ++e) {
      int idx = tid * 4 + e;
      int r = idx >> 4, c = (idx & 15) * 2;
      const float* src = &hidden[(size_t)(bx * 64 + r) * kDM + k0 + c];
      int f = r >> 4, m = r & 15;
      int lane2 = m + 16 * ((c >> 3) & 1);
      int p = ((c & 7) >> 1) + 4 * (c >> 4);
      v2h hv = {(_Float16)src[0], (_Float16)src[1]};
      *(v2h*)&af[((f * 32 + lane2) << 4) + 2 * p] = hv;
    }
    // B fill: 1024 even/odd K pairs
#pragma unroll
    for (int e = 0; e < 4; ++e) {
      int idx = tid * 4 + e;
      int n = idx >> 4, kk = (idx & 15) * 2;
      const float* src = &W[(size_t)(jcol + n) * kDM + k0 + kk];
      int lane2 = (n & 15) + 16 * ((kk >> 3) & 1);
      v2h hv = {(_Float16)src[0], (_Float16)src[1]};
      *(v2h*)&bf[(((n >> 4) * 32 + lane2) << 4) + (kk & 7) + 8 * (kk >> 4)] = hv;
    }
    __syncthreads();
    v16h a = Af[wm][lane];
    acc0 = wmma_f16(a, Bf[wn * 2][lane], acc0);
    acc1 = wmma_f16(a, Bf[wn * 2 + 1][lane], acc1);
    __syncthreads();
  }

  int n = lane & 15, mb = (lane >> 4) * 8;
#pragma unroll
  for (int vg = 0; vg < 8; ++vg) {
    int i = bx * 64 + wm * 16 + mb + vg;
    int b = i >> 12, l = i & (kL - 1);
    int j0 = jcol + wn * 32 + n;
    out[(((size_t)b * kH + (j0 >> 6)) * kL + l) * kDH + (j0 & 63)] = acc0[vg];
    int j1 = j0 + 16;
    out[(((size_t)b * kH + (j1 >> 6)) * kL + l) * kDH + (j1 & 63)] = acc1[vg];
  }
}

// ---------------------------------------------------------------------------
// Kernel 2: pooled position bias (mean of each 64x64 tile) — streams 512 MB,
// the HBM-bound leg of the whole op.
// ---------------------------------------------------------------------------
__global__ __launch_bounds__(256) void pool_bias_kernel(
    const float* __restrict__ pb, float* __restrict__ pooled) {
  int blk = blockIdx.x;                      // h * nq * nk
  int h  = blk / (kNB * kNB);
  int r  = blk % (kNB * kNB);
  int qb = r / kNB, kb = r % kNB;
  const float* base = pb + ((size_t)h * kL + qb * kBS) * kL + kb * kBS;
  float s = 0.f;
  for (int e = threadIdx.x; e < kBS * kBS; e += 256) {
    int rr = e >> 6, cc = e & 63;
    s += base[(size_t)rr * kL + cc];
  }
  __shared__ float red[256];
  red[threadIdx.x] = s;
  __syncthreads();
  for (int off = 128; off; off >>= 1) {
    if (threadIdx.x < off) red[threadIdx.x] += red[threadIdx.x + off];
    __syncthreads();
  }
  if (threadIdx.x == 0) pooled[blk] = red[0] * (1.f / 4096.f);
}

// ---------------------------------------------------------------------------
// Kernel 3: router — pool q/k, project (Wrq/Wrk), score, add bias, top-9.
// ---------------------------------------------------------------------------
__global__ __launch_bounds__(256) void router_kernel(
    const float* __restrict__ q, const float* __restrict__ k,
    const float* __restrict__ Wrq, const float* __restrict__ Wrk,
    const float* __restrict__ pooled, const float* __restrict__ bias_scale,
    int* __restrict__ sel_idx) {
  int b = blockIdx.x / kH, h = blockIdx.x % kH;
  __shared__ float qp[kNB][kDH], kp[kNB][kDH], sc[kNB][kNB];
  int tid = threadIdx.x;
  const float* qh = q + ((size_t)b * kH + h) * kL * kDH;
  const float* kh = k + ((size_t)b * kH + h) * kL * kDH;

  for (int e = tid; e < kNB * kDH; e += 256) {   // block-mean pooling
    int qb = e / kDH, d = e % kDH;
    float sq = 0.f, sk = 0.f;
    for (int r = 0; r < kBS; ++r) {
      sq += qh[(size_t)(qb * kBS + r) * kDH + d];
      sk += kh[(size_t)(qb * kBS + r) * kDH + d];
    }
    qp[qb][d] = sq * (1.f / kBS);
    kp[qb][d] = sk * (1.f / kBS);
  }
  __syncthreads();

  float tq[16], tk[16];                          // router projections
  {
    int i = 0;
    for (int e = tid; e < kNB * kDH; e += 256, ++i) {
      int rr = e / kDH, d = e % kDH;
      float sq = 0.f, sk = 0.f;
      for (int c = 0; c < kDH; ++c) {
        sq += qp[rr][c] * Wrq[d * kDH + c];
        sk += kp[rr][c] * Wrk[d * kDH + c];
      }
      tq[i] = sq; tk[i] = sk;
    }
  }
  __syncthreads();
  {
    int i = 0;
    for (int e = tid; e < kNB * kDH; e += 256, ++i) {
      int rr = e / kDH, d = e % kDH;
      qp[rr][d] = tq[i]; kp[rr][d] = tk[i];
    }
  }
  __syncthreads();

  float bsc = bias_scale[0];
  for (int e = tid; e < kNB * kNB; e += 256) {
    int qq = e / kNB, kk = e % kNB;
    float s = 0.f;
    for (int d = 0; d < kDH; ++d) s += qp[qq][d] * kp[kk][d];
    sc[qq][kk] = s * 0.125f + bsc * pooled[((size_t)h * kNB + qq) * kNB + kk];
  }
  __syncthreads();

  if (tid < kNB) {                               // top-9 by repeated argmax
    int* out = sel_idx + (((size_t)b * kH + h) * kNB + tid) * kKPQ;
    for (int j = 0; j < kKPQ; ++j) {
      float best = -3.4e38f; int bi = 0;
      for (int kk = 0; kk < kNB; ++kk)
        if (sc[tid][kk] > best) { best = sc[tid][kk]; bi = kk; }
      out[j] = bi;
      sc[tid][bi] = -3.4e38f;
    }
  }
}

// ---------------------------------------------------------------------------
// Kernel 4: flash-style block-sparse attention with per-row int8 fake-quant.
// One block per (b,h,qblock); WMMA for both S=QK^T and O+=PV.
// ---------------------------------------------------------------------------
__device__ inline void stage_tile(const float* __restrict__ src, float stage[][68]) {
  int tid = threadIdx.x;
#if ASYNC_LDS
  // 64x64 f32 = 1024 x 16B async transfers, ASYNCcnt tracked
  for (int e = tid; e < 1024; e += 256) {
    int r = e >> 4, c4 = (e & 15) * 4;
    __builtin_amdgcn_global_load_async_to_lds_b128(
        (gptr_v4i)(src + (size_t)r * kDH + c4),
        (lptr_v4i)&stage[r][c4],
        /*offset=*/0, /*cpol=*/0);
  }
  __builtin_amdgcn_s_wait_asynccnt(0);
#else
  for (int e = tid; e < kBS * kDH; e += 256) {
    int r = e >> 6, c = e & 63;
    stage[r][c] = src[(size_t)r * kDH + c];
  }
#endif
  __syncthreads();
}

// mode 0: A-layout (MxK, rows of src are M)      [q, probs]
// mode 1: B-layout from transposed src (KxN via B[k=c][n=r])   [k block]
// mode 2: B-layout direct (KxN via B[k=r][n=c])  [v block]
__device__ void load_quant_tile(const float* __restrict__ src, _Float16* dst,
                                int mode, float stage[][68], float* rowscale) {
  stage_tile(src, stage);
  int tid = threadIdx.x;
  if (tid < kBS) {
    float m = 0.f;
    for (int c = 0; c < kDH; ++c) m = fmaxf(m, fabsf(stage[tid][c]));
    rowscale[tid] = fmaxf(m, 1e-8f) * (1.f / 127.f);
  }
  __syncthreads();
  for (int e = tid; e < kBS * kDH; e += 256) {
    int r = e >> 6, c = e & 63;
    float qv = fq(stage[r][c], rowscale[r]);
    int di = (mode == 0) ? a_frag_index(r, c)
           : (mode == 1) ? b_frag_index(c, r)
                         : b_frag_index(r, c);
    dst[di] = (_Float16)qv;
  }
  __syncthreads();
}

__global__ __launch_bounds__(256) void sparse_attn_kernel(
    const float* __restrict__ q, const float* __restrict__ k,
    const float* __restrict__ v, const float* __restrict__ pb,
    const int* __restrict__ sel_idx, float* __restrict__ sparse_out) {
  __shared__ v16h Qf[2][4][32];   // [kstep][M-strip][lane]
  __shared__ v16h Kf[2][4][32];   // [kstep][N-group][lane]
  __shared__ v16h Vf[2][4][32];
  __shared__ v16h Pf[2][4][32];
  __shared__ float sS[64][68];
  __shared__ float rowscale[64], rmax[64], rsum[64], rfac[64];

  int blk = blockIdx.x;
  int qb = blk % kNB;
  int h  = (blk / kNB) % kH;
  int b  = blk / (kNB * kH);
  int tid = threadIdx.x;
  int wave = tid >> 5, lane = tid & 31;
  int wm = wave >> 1, wn = wave & 1;
  int n_ = lane & 15, mb_ = (lane >> 4) * 8;

  const size_t headoff = ((size_t)b * kH + h) * kL * kDH;
  load_quant_tile(q + headoff + (size_t)qb * kBS * kDH, (_Float16*)Qf, 0, sS, rowscale);

  if (tid < kBS) { rmax[tid] = -3.0e38f; rsum[tid] = 0.f; }
  __syncthreads();

  v8f o0 = {}; v8f o1 = {};
  const int* sel = sel_idx + (((size_t)b * kH + h) * kNB + qb) * kKPQ;

  for (int j = 0; j < kKPQ; ++j) {
    int kb = sel[j];
    load_quant_tile(k + headoff + (size_t)kb * kBS * kDH, (_Float16*)Kf, 1, sS, rowscale);
    load_quant_tile(v + headoff + (size_t)kb * kBS * kDH, (_Float16*)Vf, 2, sS, rowscale);

    // S = q @ k^T (64x64) + position-bias block
    v8f s0 = {}; v8f s1 = {};
#pragma unroll
    for (int ks = 0; ks < 2; ++ks) {
      v16h a = Qf[ks][wm][lane];
      s0 = wmma_f16(a, Kf[ks][wn * 2][lane], s0);
      s1 = wmma_f16(a, Kf[ks][wn * 2 + 1][lane], s1);
    }
#pragma unroll
    for (int vg = 0; vg < 8; ++vg) {
      int m = wm * 16 + mb_ + vg;
      const float* pbrow = pb + ((size_t)h * kL + qb * kBS + m) * kL + kb * kBS;
      s0[vg] += pbrow[wn * 32 + n_];
      s1[vg] += pbrow[wn * 32 + 16 + n_];
    }
    store_c16x16(&sS[wm * 16][wn * 32], 68, s0);
    store_c16x16(&sS[wm * 16][wn * 32 + 16], 68, s1);
    __syncthreads();

    // online softmax over this 64-column chunk
    if (tid < kBS) {
      float bm = -3.0e38f;
      for (int c = 0; c < kBS; ++c) bm = fmaxf(bm, sS[tid][c]);
      float nm  = fmaxf(rmax[tid], bm);
      float fac = __expf(rmax[tid] - nm);
      float s = 0.f, mp = 0.f;
      for (int c = 0; c < kBS; ++c) {
        float e = __expf(sS[tid][c] - nm);
        sS[tid][c] = e; s += e; mp = fmaxf(mp, e);
      }
      rsum[tid] = rsum[tid] * fac + s;
      rmax[tid] = nm; rfac[tid] = fac;
      rowscale[tid] = fmaxf(mp, 1e-8f) * (1.f / 127.f);
    }
    __syncthreads();

    // rescale running accumulator, fake-quantize probabilities into Pf
#pragma unroll
    for (int vg = 0; vg < 8; ++vg) {
      float f = rfac[wm * 16 + mb_ + vg];
      o0[vg] *= f; o1[vg] *= f;
    }
    {
      _Float16* pf = (_Float16*)Pf;
      for (int e = tid; e < kBS * kBS; e += 256) {
        int r = e >> 6, c = e & 63;
        pf[a_frag_index(r, c)] = (_Float16)fq(sS[r][c], rowscale[r]);
      }
    }
    __syncthreads();

    // O += P @ V
#pragma unroll
    for (int ks = 0; ks < 2; ++ks) {
      v16h a = Pf[ks][wm][lane];
      o0 = wmma_f16(a, Vf[ks][wn * 2][lane], o0);
      o1 = wmma_f16(a, Vf[ks][wn * 2 + 1][lane], o1);
    }
    __syncthreads();
  }

  // normalize by softmax denominator, write out
#pragma unroll
  for (int vg = 0; vg < 8; ++vg) {
    int m = wm * 16 + mb_ + vg;
    float inv = 1.f / rsum[m];
    float* orow = sparse_out + headoff + (size_t)(qb * kBS + m) * kDH;
    orow[wn * 32 + n_]      = o0[vg] * inv;
    orow[wn * 32 + 16 + n_] = o1[vg] * inv;
  }
}

// ---------------------------------------------------------------------------
// Kernel 5: linear-attention branch.  One block per (b,h).
// ---------------------------------------------------------------------------
__global__ __launch_bounds__(256) void linear_attn_kernel(
    const float* __restrict__ q, const float* __restrict__ k,
    const float* __restrict__ v, float* __restrict__ lin_out) {
  int b = blockIdx.x / kH, h = blockIdx.x % kH;
  __shared__ float pk[kBS][kDH + 1], vv[kBS][kDH + 1];
  __shared__ float kvs[kDH][kDH + 1], ksum[kDH], den[kBS];
  int tid = threadIdx.x;
  const size_t headoff = ((size_t)b * kH + h) * kL * kDH;
  const float* qh = q + headoff;
  const float* kh = k + headoff;
  const float* vhh = v + headoff;

  const int d0 = (tid * 16) >> 6;
  const int m0 = (tid * 16) & 63;
  float acc[16];
#pragma unroll
  for (int i = 0; i < 16; ++i) acc[i] = 0.f;
  float ks = 0.f;

  for (int t0 = 0; t0 < kL; t0 += kBS) {
    for (int e = tid; e < kBS * kDH; e += 256) {
      int r = e >> 6, c = e & 63;
      pk[r][c] = phi_fn(kh[(size_t)(t0 + r) * kDH + c]);
      vv[r][c] = vhh[(size_t)(t0 + r) * kDH + c];
    }
    __syncthreads();
    for (int r = 0; r < kBS; ++r) {
      float pkd = pk[r][d0];
#pragma unroll
      for (int i = 0; i < 16; ++i) acc[i] += pkd * vv[r][m0 + i];
    }
    if (tid < kDH) {
      float s = 0.f;
      for (int r = 0; r < kBS; ++r) s += pk[r][tid];
      ks += s;
    }
    __syncthreads();
  }
#pragma unroll
  for (int i = 0; i < 16; ++i) kvs[d0][m0 + i] = acc[i];
  if (tid < kDH) ksum[tid] = ks;
  __syncthreads();

  for (int t0 = 0; t0 < kL; t0 += kBS) {
    for (int e = tid; e < kBS * kDH; e += 256) {
      int r = e >> 6, c = e & 63;
      pk[r][c] = phi_fn(qh[(size_t)(t0 + r) * kDH + c]);
    }
    __syncthreads();
    if (tid < kBS) {
      float s = 0.f;
      for (int d = 0; d < kDH; ++d) s += pk[tid][d] * ksum[d];
      den[tid] = s + 1e-6f;
    }
    __syncthreads();
    for (int e = tid; e < kBS * kDH; e += 256) {
      int r = e >> 6, m = e & 63;
      float s = 0.f;
      for (int d = 0; d < kDH; ++d) s += pk[r][d] * kvs[d][m];
      lin_out[headoff + (size_t)(t0 + r) * kDH + m] = s / den[r];
    }
    __syncthreads();
  }
}

// ---------------------------------------------------------------------------
// Kernel 6: blend (sigmoid gate) + output projection  out = blend @ Wo.T
// ---------------------------------------------------------------------------
__global__ __launch_bounds__(256) void out_proj_kernel(
    const float* __restrict__ sparse, const float* __restrict__ lin,
    const float* __restrict__ alpha_logits, const float* __restrict__ Wo,
    float* __restrict__ out) {
  __shared__ v16h Af[4][32];
  __shared__ v16h Bf[4][32];
  _Float16* af = (_Float16*)Af;
  _Float16* bf = (_Float16*)Bf;
  const int bx = blockIdx.x;
  const int by = blockIdx.y;
  const int tid = threadIdx.x;
  const int wave = tid >> 5, lane = tid & 31;
  const int wm = wave >> 1, wn = wave & 1;

  v8f acc0 = {}; v8f acc1 = {};

  for (int k0 = 0; k0 < kDM; k0 += 32) {
#pragma unroll
    for (int e = 0; e < 4; ++e) {
      int idx = tid * 4 + e;
      int r = idx >> 4, cc = (idx & 15) * 2;
      int c = k0 + cc;
      int hh = c >> 6, dd = c & 63;
      int i = bx * 64 + r;
      int b = i >> 12, l = i & (kL - 1);
      size_t off = (((size_t)b * kH + hh) * kL + l) * kDH + dd;
      float al = 1.f / (1.f + __expf(-alpha_logits[hh]));
      float x0 = al * sparse[off] + (1.f - al) * lin[off];
      float x1 = al * sparse[off + 1] + (1.f - al) * lin[off + 1];
      int f = r >> 4, m = r & 15;
      int lane2 = m + 16 * ((cc >> 3) & 1);
      int p = ((cc & 7) >> 1) + 4 * (cc >> 4);
      v2h hv = {(_Float16)x0, (_Float16)x1};
      *(v2h*)&af[((f * 32 + lane2) << 4) + 2 * p] = hv;
    }
#pragma unroll
    for (int e = 0; e < 4; ++e) {
      int idx = tid * 4 + e;
      int n = idx >> 4, kk = (idx & 15) * 2;
      const float* src = &Wo[(size_t)(by * 64 + n) * kDM + k0 + kk];
      int lane2 = (n & 15) + 16 * ((kk >> 3) & 1);
      v2h hv = {(_Float16)src[0], (_Float16)src[1]};
      *(v2h*)&bf[(((n >> 4) * 32 + lane2) << 4) + (kk & 7) + 8 * (kk >> 4)] = hv;
    }
    __syncthreads();
    v16h a = Af[wm][lane];
    acc0 = wmma_f16(a, Bf[wn * 2][lane], acc0);
    acc1 = wmma_f16(a, Bf[wn * 2 + 1][lane], acc1);
    __syncthreads();
  }

  int n = lane & 15, mb = (lane >> 4) * 8;
#pragma unroll
  for (int vg = 0; vg < 8; ++vg) {
    int i = bx * 64 + wm * 16 + mb + vg;
    int b = i >> 12, l = i & (kL - 1);
    int j = by * 64 + wn * 32 + n;
    out[((size_t)b * kL + l) * kDM + j]      = acc0[vg];
    out[((size_t)b * kL + l) * kDM + j + 16] = acc1[vg];
  }
}

// ---------------------------------------------------------------------------
// Launch
// ---------------------------------------------------------------------------
extern "C" void kernel_launch(void* const* d_in, const int* in_sizes, int n_in,
                              void* d_out, int out_size, void* d_ws, size_t ws_size,
                              hipStream_t stream) {
  const float* hidden       = (const float*)d_in[0];
  const float* pb           = (const float*)d_in[1];
  const float* Wq           = (const float*)d_in[2];
  const float* Wk           = (const float*)d_in[3];
  const float* Wv           = (const float*)d_in[4];
  const float* Wo           = (const float*)d_in[5];
  const float* Wrq          = (const float*)d_in[6];
  const float* Wrk          = (const float*)d_in[7];
  const float* bias_scale   = (const float*)d_in[8];
  const float* alpha_logits = (const float*)d_in[9];

  const size_t TEN = (size_t)kB * kH * kL * kDH;   // 4,194,304 floats
  float* q      = (float*)d_ws;
  float* k      = q + TEN;
  float* v      = k + TEN;
  float* sp     = v + TEN;
  float* lin    = sp + TEN;
  float* pooled = lin + TEN;
  int*   sel    = (int*)(pooled + (size_t)kH * kNB * kNB);

  qkv_proj_kernel<<<dim3(kB * kL / 64, 3 * kDM / 64), 256, 0, stream>>>(
      hidden, Wq, Wk, Wv, q, k, v);
  pool_bias_kernel<<<kH * kNB * kNB, 256, 0, stream>>>(pb, pooled);
  router_kernel<<<kB * kH, 256, 0, stream>>>(q, k, Wrq, Wrk, pooled, bias_scale, sel);
  sparse_attn_kernel<<<kB * kH * kNB, 256, 0, stream>>>(q, k, v, pb, sel, sp);
  linear_attn_kernel<<<kB * kH, 256, 0, stream>>>(q, k, v, lin);
  out_proj_kernel<<<dim3(kB * kL / 64, kDM / 64), 256, 0, stream>>>(
      sp, lin, alpha_logits, Wo, (float*)d_out);
}
